// DecoderLayer_19396072309085
// MI455X (gfx1250) — matrix operand order repaired and checked by reading, not compiled, not run on hardware
//
#include <hip/hip_runtime.h>

// ---------------------------------------------------------------------------
// Problem constants (match reference)
// ---------------------------------------------------------------------------
#define BB 8
#define SS 512
#define TT 512
#define DD 1024
#define HH 16
#define KK 64
#define FF 4096
#define BS (BB * SS)                          // 4096 rows
#define BHST ((long long)BB * HH * SS * TT)   // 33,554,432

typedef __attribute__((ext_vector_type(16))) __bf16          v16bf;
typedef __attribute__((ext_vector_type(8)))  float           v8f;
typedef __attribute__((__vector_size__(4 * sizeof(int)))) int nv4i;  // native int4

struct Q2 { uint4 a, b; };   // 32 bytes == one v16bf fragment

#if defined(__HIP_DEVICE_COMPILE__) && __has_builtin(__builtin_amdgcn_global_load_async_to_lds_b128)
#define HAS_ASYNC_LDS 1
#else
#define HAS_ASYNC_LDS 0
#endif

#if defined(__HIP_DEVICE_COMPILE__) && __has_builtin(__builtin_amdgcn_s_wait_asynccnt)
#define WAIT_ASYNC() __builtin_amdgcn_s_wait_asynccnt(0)
#else
#define WAIT_ASYNC() asm volatile("s_wait_asynccnt 0x0" ::: "memory")
#endif

__device__ __forceinline__ unsigned short f2bf(float f) {
    unsigned int u = __float_as_uint(f);
    u += 0x7FFFu + ((u >> 16) & 1u);      // round-to-nearest-even
    return (unsigned short)(u >> 16);
}

// ---------------------------------------------------------------------------
// fp32 -> bf16 conversion
// ---------------------------------------------------------------------------
__global__ void k_f32_to_bf16(const float* __restrict__ in,
                              unsigned short* __restrict__ out, long long n) {
    long long i = (long long)blockIdx.x * 256 + threadIdx.x;
    if (i < n) out[i] = f2bf(in[i]);
}

// ---------------------------------------------------------------------------
// Generic batched strided bf16 WMMA GEMM:  C = alpha * A(MxK) @ B(KxN) + bias
// A row-major (row stride lda, contiguous rows).  B element (k,n) at
// k*ldbk + n*ldbn.  Batch (b,h) offsets via per-dim strides.  Optional ReLU.
// Block = 256 threads (8 wave32), tile = 64(M) x 64(N), K staged 64 at a time.
// A tile staged row-major via async global->LDS b128.  B tile staged
// TRANSPOSED (Bt[n][k]):
//   - ldbk==1 (scores QK^T): columns contiguous in memory -> async b128 too
//   - ldbn==1 (proj/FFN/ctx): vectorized b128 read + LDS scatter (transpose)
// Fragments then load as contiguous ds_load_b128 pairs.
// ---------------------------------------------------------------------------
__global__ __launch_bounds__(256) void k_gemm_bf16(
    const unsigned short* __restrict__ A, long long aB, long long aH, int lda,
    const unsigned short* __restrict__ Bm, long long bB, long long bH,
    long long ldbk, long long ldbn,
    const float* __restrict__ bias,
    float* __restrict__ Cf, unsigned short* __restrict__ Cbf,
    long long cB, long long cH, int ldc,
    int Kd, int Hn, float alpha, int relu) {

    // stride 72 ushorts = 144B: 16B aligned sub-rows, conflict-free frag reads
    __shared__ alignas(16) unsigned short As[64][72];
    __shared__ alignas(16) unsigned short Bt[64][72];   // transposed: [n][k]

    const int z  = blockIdx.z;
    const int bb = z / Hn;
    const int hh = z % Hn;
    A  += bb * aB + hh * aH;
    Bm += bb * bB + hh * bH;
    const long long coff = bb * cB + hh * cH;

    const int m0 = blockIdx.y * 64;
    const int n0 = blockIdx.x * 64;
    const int tid   = threadIdx.x;
    const int lane  = tid & 31;
    const int w     = tid >> 5;
    const int wm    = w & 3;
    const int wn    = w >> 2;
    const int khalf = lane >> 4;
    const int nl    = lane & 15;
    const int arow  = wm * 16 + nl;
    const int n1    = wn * 32 + nl;

    v8f acc0 = {};
    v8f acc1 = {};

    for (int kk = 0; kk < Kd; kk += 64) {
        // ---- stage A tile 64x64 (8 ushorts = 16B per transfer) ----
#pragma unroll
        for (int i = 0; i < 2; ++i) {
            int e = (tid + i * 256) * 8;
            int r = e >> 6, c = e & 63;
            const unsigned short* gp =
                A + (long long)(m0 + r) * lda + (kk + c);
#if HAS_ASYNC_LDS
            __builtin_amdgcn_global_load_async_to_lds_b128(
                (nv4i*)gp, (nv4i*)&As[r][c], 0, 0);
#else
            *(uint4*)&As[r][c] = *(const uint4*)gp;
#endif
        }
        // ---- stage B tile 64x64 transposed into Bt[n][k] ----
        if (ldbk == 1) {
            // B columns contiguous in memory (QK^T): direct async copy
#pragma unroll
            for (int i = 0; i < 2; ++i) {
                int e = (tid + i * 256) * 8;
                int r = e & 63, c = e >> 6;           // r = k base, c = n
                const unsigned short* gp =
                    Bm + (long long)(n0 + c) * ldbn + (kk + r);
#if HAS_ASYNC_LDS
                __builtin_amdgcn_global_load_async_to_lds_b128(
                    (nv4i*)gp, (nv4i*)&Bt[c][r], 0, 0);
#else
                *(uint4*)&Bt[c][r] = *(const uint4*)gp;
#endif
            }
        } else if (ldbn == 1) {
            // B rows contiguous: vector read + LDS scatter (transpose)
#pragma unroll
            for (int i = 0; i < 2; ++i) {
                int e = (tid + i * 256) * 8;
                int r = e >> 6, c = e & 63;           // r = k, c = n base
                uint4 qv = *(const uint4*)(Bm + (long long)(kk + r) * ldbk +
                                           (n0 + c));
                const unsigned short* us = (const unsigned short*)&qv;
#pragma unroll
                for (int j = 0; j < 8; ++j) Bt[c + j][r] = us[j];
            }
        } else {
#pragma unroll
            for (int i = 0; i < 16; ++i) {
                int e = tid + i * 256;
                int r = e >> 6, c = e & 63;
                Bt[c][r] = Bm[(long long)(kk + r) * ldbk +
                              (long long)(n0 + c) * ldbn];
            }
        }
#if HAS_ASYNC_LDS
        WAIT_ASYNC();
#endif
        __syncthreads();

        // ---- 2 sub-chunks of K=32, 2 WMMAs each ----
#pragma unroll
        for (int kc = 0; kc < 64; kc += 32) {
            Q2 qa  = { *(const uint4*)&As[arow][kc + khalf * 16],
                       *(const uint4*)&As[arow][kc + khalf * 16 + 8] };
            Q2 qb0 = { *(const uint4*)&Bt[n1][kc + khalf * 16],
                       *(const uint4*)&Bt[n1][kc + khalf * 16 + 8] };
            Q2 qb1 = { *(const uint4*)&Bt[n1 + 16][kc + khalf * 16],
                       *(const uint4*)&Bt[n1 + 16][kc + khalf * 16 + 8] };
            v16bf av = __builtin_bit_cast(v16bf, qa);
            acc0 = __builtin_amdgcn_wmma_f32_16x16x32_bf16(
                false, av, false, __builtin_bit_cast(v16bf, qb0), (short)0,
                acc0, false, false);
            acc1 = __builtin_amdgcn_wmma_f32_16x16x32_bf16(
                false, av, false, __builtin_bit_cast(v16bf, qb1), (short)0,
                acc1, false, false);
        }
        __syncthreads();
    }

    // ---- epilogue: C layout VGPR r -> M = r + 8*khalf, N = lane&15 ----
    const int gn0 = n0 + wn * 32 + nl;
    const int gn1 = gn0 + 16;
    const float b0 = bias ? bias[gn0] : 0.f;
    const float b1 = bias ? bias[gn1] : 0.f;
#pragma unroll
    for (int r = 0; r < 8; ++r) {
        int gm = m0 + wm * 16 + khalf * 8 + r;
        float v0 = acc0[r] * alpha + b0;
        float v1 = acc1[r] * alpha + b1;
        if (relu) { v0 = fmaxf(v0, 0.f); v1 = fmaxf(v1, 0.f); }
        long long i0 = coff + (long long)gm * ldc + gn0;
        long long i1 = coff + (long long)gm * ldc + gn1;
        if (Cf)  { Cf[i0] = v0;        Cf[i1] = v1; }
        if (Cbf) { Cbf[i0] = f2bf(v0); Cbf[i1] = f2bf(v1); }
    }
}

// ---------------------------------------------------------------------------
// Row softmax, in-place on fp32 scores, bf16 copy out.  One 128-thread block
// per row of length T (=512 -> 4 elems/thread).  Optional causal mask.
// ---------------------------------------------------------------------------
__global__ __launch_bounds__(128) void k_softmax(float* __restrict__ sc,
                                                 unsigned short* __restrict__ obf,
                                                 int T, int S, int causal) {
    __shared__ float red[128];
    const int row  = blockIdx.x;
    const int tid  = threadIdx.x;
    const int s    = row % S;
    const long long base = (long long)row * T;

    float v[4];
    float mx = -3.4e38f;
#pragma unroll
    for (int j = 0; j < 4; ++j) {
        int t = tid + j * 128;
        float x = sc[base + t];
        if (causal && t > s) x = -1e9f;
        v[j] = x;
        mx = fmaxf(mx, x);
    }
    red[tid] = mx;
    __syncthreads();
    for (int off = 64; off > 0; off >>= 1) {
        if (tid < off) red[tid] = fmaxf(red[tid], red[tid + off]);
        __syncthreads();
    }
    mx = red[0];
    __syncthreads();

    float sum = 0.f;
#pragma unroll
    for (int j = 0; j < 4; ++j) {
        v[j] = __expf(v[j] - mx);   // masked lanes underflow to 0
        sum += v[j];
    }
    red[tid] = sum;
    __syncthreads();
    for (int off = 64; off > 0; off >>= 1) {
        if (tid < off) red[tid] += red[tid + off];
        __syncthreads();
    }
    const float inv = 1.f / red[0];
#pragma unroll
    for (int j = 0; j < 4; ++j) {
        int t = tid + j * 128;
        float p = v[j] * inv;
        sc[base + t]  = p;
        obf[base + t] = f2bf(p);
    }
}

// ---------------------------------------------------------------------------
// Residual-add + LayerNorm over D=1024.  256 threads/row, 4 elems/thread.
// Emits fp32 (for next residual) and optional bf16 (for next GEMM).
// ---------------------------------------------------------------------------
__global__ __launch_bounds__(256) void k_ln(const float* __restrict__ a,
                                            const float* __restrict__ b,
                                            const float* __restrict__ g,
                                            const float* __restrict__ be,
                                            float* __restrict__ of,
                                            unsigned short* __restrict__ obf,
                                            int Dd) {
    __shared__ float r1[256];
    __shared__ float r2[256];
    const int row = blockIdx.x;
    const int tid = threadIdx.x;
    const long long base = (long long)row * Dd;

    float v[4];
    float s1 = 0.f, s2 = 0.f;
#pragma unroll
    for (int j = 0; j < 4; ++j) {
        int i = tid + j * 256;
        float x = a[base + i] + b[base + i];
        v[j] = x;
        s1 += x;
        s2 += x * x;
    }
    r1[tid] = s1; r2[tid] = s2;
    __syncthreads();
    for (int off = 128; off > 0; off >>= 1) {
        if (tid < off) { r1[tid] += r1[tid + off]; r2[tid] += r2[tid + off]; }
        __syncthreads();
    }
    const float mean = r1[0] / Dd;
    const float var  = r2[0] / Dd - mean * mean;
    const float rstd = rsqrtf(var + 1e-6f);
#pragma unroll
    for (int j = 0; j < 4; ++j) {
        int i = tid + j * 256;
        float y = (v[j] - mean) * rstd * g[i] + be[i];
        of[base + i] = y;
        if (obf) obf[base + i] = f2bf(y);
    }
}

// ---------------------------------------------------------------------------
// Orchestration
// ---------------------------------------------------------------------------
extern "C" void kernel_launch(void* const* d_in, const int* in_sizes, int n_in,
                              void* d_out, int out_size, void* d_ws, size_t ws_size,
                              hipStream_t stream) {
    (void)in_sizes; (void)n_in; (void)out_size; (void)ws_size;

    // ---- inputs (setup_inputs order) ----
    const float* x    = (const float*)d_in[0];
    const float* enc  = (const float*)d_in[1];
    // d_in[2] = look_ahead_mask (causal tril) -> handled via causal flag
    const float* wq1 = (const float*)d_in[3];  const float* bq1 = (const float*)d_in[4];
    const float* wk1 = (const float*)d_in[5];  const float* bk1 = (const float*)d_in[6];
    const float* wv1 = (const float*)d_in[7];  const float* bv1 = (const float*)d_in[8];
    const float* wo1 = (const float*)d_in[9];  const float* bo1 = (const float*)d_in[10];
    const float* wq2 = (const float*)d_in[11]; const float* bq2 = (const float*)d_in[12];
    const float* wk2 = (const float*)d_in[13]; const float* bk2 = (const float*)d_in[14];
    const float* wv2 = (const float*)d_in[15]; const float* bv2 = (const float*)d_in[16];
    const float* wo2 = (const float*)d_in[17]; const float* bo2 = (const float*)d_in[18];
    const float* g1  = (const float*)d_in[19]; const float* be1 = (const float*)d_in[20];
    const float* g2  = (const float*)d_in[21]; const float* be2 = (const float*)d_in[22];
    const float* g3  = (const float*)d_in[23]; const float* be3 = (const float*)d_in[24];
    const float* wf1 = (const float*)d_in[25]; const float* bf1 = (const float*)d_in[26];
    const float* wf2 = (const float*)d_in[27]; const float* bf2 = (const float*)d_in[28];

    // ---- output regions ----
    float* out_dec   = (float*)d_out;                       // [B,S,D]
    float* out_attn1 = out_dec + (long long)BS * DD;        // [B,H,S,S]
    float* out_attn2 = out_attn1 + BHST;                    // [B,H,S,T]

    // ---- workspace carve-out ----
    char* ws = (char*)d_ws;
    size_t off = 0;
    auto alloc = [&](size_t bytes) {
        size_t o = off;
        off = (off + bytes + 255) & ~(size_t)255;
        return (void*)(ws + o);
    };
    const size_t U = sizeof(unsigned short);
    unsigned short* x_bf    = (unsigned short*)alloc((size_t)BS * DD * U);
    unsigned short* enc_bf  = (unsigned short*)alloc((size_t)BB * TT * DD * U);
    unsigned short* wq1_bf  = (unsigned short*)alloc((size_t)DD * DD * U);
    unsigned short* wk1_bf  = (unsigned short*)alloc((size_t)DD * DD * U);
    unsigned short* wv1_bf  = (unsigned short*)alloc((size_t)DD * DD * U);
    unsigned short* wo1_bf  = (unsigned short*)alloc((size_t)DD * DD * U);
    unsigned short* wq2_bf  = (unsigned short*)alloc((size_t)DD * DD * U);
    unsigned short* wk2_bf  = (unsigned short*)alloc((size_t)DD * DD * U);
    unsigned short* wv2_bf  = (unsigned short*)alloc((size_t)DD * DD * U);
    unsigned short* wo2_bf  = (unsigned short*)alloc((size_t)DD * DD * U);
    unsigned short* wf1_bf  = (unsigned short*)alloc((size_t)DD * FF * U);
    unsigned short* wf2_bf  = (unsigned short*)alloc((size_t)FF * DD * U);
    unsigned short* q_bf    = (unsigned short*)alloc((size_t)BS * DD * U);
    unsigned short* k_bf    = (unsigned short*)alloc((size_t)BS * DD * U);
    unsigned short* v_bf    = (unsigned short*)alloc((size_t)BS * DD * U);
    unsigned short* attn_bf = (unsigned short*)alloc((size_t)BHST * U);
    unsigned short* ctx_bf  = (unsigned short*)alloc((size_t)BS * DD * U);
    unsigned short* out1_bf = (unsigned short*)alloc((size_t)BS * DD * U);
    unsigned short* out2_bf = (unsigned short*)alloc((size_t)BS * DD * U);
    unsigned short* h_bf    = (unsigned short*)alloc((size_t)BS * FF * U);
    float* attnout_f = (float*)alloc((size_t)BS * DD * 4);
    float* out1_f    = (float*)alloc((size_t)BS * DD * 4);
    float* out2_f    = (float*)alloc((size_t)BS * DD * 4);
    float* ffn_f     = (float*)alloc((size_t)BS * DD * 4);

    auto conv = [&](const float* src, unsigned short* dst, long long n) {
        int blocks = (int)((n + 255) / 256);
        k_f32_to_bf16<<<blocks, 256, 0, stream>>>(src, dst, n);
    };
    auto gemm = [&](const unsigned short* A, long long aB, long long aH, int lda,
                    const unsigned short* Bm, long long bB, long long bH,
                    long long ldbk, long long ldbn,
                    const float* bias, float* Cf, unsigned short* Cbf,
                    long long cB, long long cH, int ldc,
                    int M, int N, int Kd, int batches, int Hn,
                    float alpha, int relu) {
        dim3 grid(N / 64, M / 64, batches);
        k_gemm_bf16<<<grid, 256, 0, stream>>>(A, aB, aH, lda, Bm, bB, bH, ldbk,
                                              ldbn, bias, Cf, Cbf, cB, cH, ldc,
                                              Kd, Hn, alpha, relu);
    };

    const float SC = 0.125f;  // 1/sqrt(64)

    // ---- bf16 conversions ----
    conv(x,   x_bf,   (long long)BS * DD);
    conv(enc, enc_bf, (long long)BB * TT * DD);
    conv(wq1, wq1_bf, (long long)DD * DD); conv(wk1, wk1_bf, (long long)DD * DD);
    conv(wv1, wv1_bf, (long long)DD * DD); conv(wo1, wo1_bf, (long long)DD * DD);
    conv(wq2, wq2_bf, (long long)DD * DD); conv(wk2, wk2_bf, (long long)DD * DD);
    conv(wv2, wv2_bf, (long long)DD * DD); conv(wo2, wo2_bf, (long long)DD * DD);
    conv(wf1, wf1_bf, (long long)DD * FF); conv(wf2, wf2_bf, (long long)FF * DD);

    // ================= MHA1 (self, causal) =================
    gemm(x_bf, 0,0, DD, wq1_bf, 0,0, DD,1, bq1, nullptr, q_bf, 0,0, DD,
         BS, DD, DD, 1, 1, 1.f, 0);
    gemm(x_bf, 0,0, DD, wk1_bf, 0,0, DD,1, bk1, nullptr, k_bf, 0,0, DD,
         BS, DD, DD, 1, 1, 1.f, 0);
    gemm(x_bf, 0,0, DD, wv1_bf, 0,0, DD,1, bv1, nullptr, v_bf, 0,0, DD,
         BS, DD, DD, 1, 1, 1.f, 0);
    // scores[b,h] = q[b,:,h,:] @ k[b,:,h,:]^T * SC  -> d_out attn1 region
    gemm(q_bf, (long long)SS * DD, KK, DD,
         k_bf, (long long)TT * DD, KK, 1, DD,
         nullptr, out_attn1, nullptr,
         (long long)HH * SS * TT, (long long)SS * TT, SS,
         SS, SS, KK, BB * HH, HH, SC, 0);
    k_softmax<<<BB * HH * SS, 128, 0, stream>>>(out_attn1, attn_bf, SS, SS, 1);
    // ctx[b,h] = attn @ v[b,:,h,:]
    gemm(attn_bf, (long long)HH * SS * TT, (long long)SS * TT, SS,
         v_bf, (long long)TT * DD, KK, DD, 1,
         nullptr, nullptr, ctx_bf,
         (long long)SS * DD, KK, DD,
         SS, KK, SS, BB * HH, HH, 1.f, 0);
    // output projection: ctx[BS, H*K] @ wo1[H*K, D] + bo1
    gemm(ctx_bf, 0,0, DD, wo1_bf, 0,0, DD,1, bo1, attnout_f, nullptr, 0,0, DD,
         BS, DD, DD, 1, 1, 1.f, 0);
    k_ln<<<BS, 256, 0, stream>>>(attnout_f, x, g1, be1, out1_f, out1_bf, DD);

    // ================= MHA2 (cross, no mask) =================
    gemm(out1_bf, 0,0, DD, wq2_bf, 0,0, DD,1, bq2, nullptr, q_bf, 0,0, DD,
         BS, DD, DD, 1, 1, 1.f, 0);
    gemm(enc_bf, 0,0, DD, wk2_bf, 0,0, DD,1, bk2, nullptr, k_bf, 0,0, DD,
         BB * TT, DD, DD, 1, 1, 1.f, 0);
    gemm(enc_bf, 0,0, DD, wv2_bf, 0,0, DD,1, bv2, nullptr, v_bf, 0,0, DD,
         BB * TT, DD, DD, 1, 1, 1.f, 0);
    gemm(q_bf, (long long)SS * DD, KK, DD,
         k_bf, (long long)TT * DD, KK, 1, DD,
         nullptr, out_attn2, nullptr,
         (long long)HH * SS * TT, (long long)SS * TT, TT,
         SS, TT, KK, BB * HH, HH, SC, 0);
    k_softmax<<<BB * HH * SS, 128, 0, stream>>>(out_attn2, attn_bf, TT, SS, 0);
    gemm(attn_bf, (long long)HH * SS * TT, (long long)SS * TT, TT,
         v_bf, (long long)TT * DD, KK, DD, 1,
         nullptr, nullptr, ctx_bf,
         (long long)SS * DD, KK, DD,
         SS, KK, TT, BB * HH, HH, 1.f, 0);
    gemm(ctx_bf, 0,0, DD, wo2_bf, 0,0, DD,1, bo2, attnout_f, nullptr, 0,0, DD,
         BS, DD, DD, 1, 1, 1.f, 0);
    k_ln<<<BS, 256, 0, stream>>>(attnout_f, out1_f, g2, be2, out2_f, out2_bf, DD);

    // ================= FFN =================
    gemm(out2_bf, 0,0, DD, wf1_bf, 0,0, FF,1, bf1, nullptr, h_bf, 0,0, FF,
         BS, FF, DD, 1, 1, 1.f, 1 /*relu*/);
    gemm(h_bf, 0,0, FF, wf2_bf, 0,0, DD,1, bf2, ffn_f, nullptr, 0,0, DD,
         BS, DD, FF, 1, 1, 1.f, 0);
    k_ln<<<BS, 256, 0, stream>>>(ffn_f, out2_f, g3, be3, out_dec, nullptr, DD);
}
